// ShiftedWindowAttention_17858474017369
// MI455X (gfx1250) — matrix operand order repaired and checked by reading, not compiled, not run on hardware
//
#include <hip/hip_runtime.h>

// ---------------------------------------------------------------------------
// Fused windowed attention for MI455X (gfx1250, wave32, WMMA bf16).
// One workgroup per batch window (B=8192). All intermediates live in LDS:
//   x(bf16) -> qkv GEMM -> Q,K in LDS, V stored transposed (Vt) ->
//   per-head S=QK^T (1 WMMA/tile, K=hd=32) -> softmax -> O=PV -> proj GEMM.
// Weights pre-transposed+converted to bf16 in workspace (L2-resident).
// A-fragments follow the CDNA5 16-bit A layout (ISA 7.12.2): per lane the
// 16 values are two 8-element K-chunks: [k0+hi*8, +8) and [k0+16+hi*8, +8).
// B-fragments use the same interleaved 8-element K-chunks (column source rows).
// ---------------------------------------------------------------------------

typedef __attribute__((ext_vector_type(16))) __bf16 v16bf;
typedef __attribute__((ext_vector_type(8)))  __bf16 v8bf;
typedef __attribute__((ext_vector_type(8)))  float  v8f;

#define NHEADS 12
#define HD     32
#define SEQ    49
#define CH     384
#define CH3    1152

// LDS pitches (elements). bf16 pitches all multiples of 16 (alignment of fragment loads).
#define XS_P  400   // bf16 x staging, aliased by Ob in phase C/D
#define QKV_P 784   // bf16 Q (cols 0..383) | K (cols 384..767)
#define VT_P  80    // bf16 Vt[h*32+d][seq]
#define SF_P  68    // f32 scores
#define PB_P  80    // bf16 probabilities

#define O_XS   0
#define O_QKV  (O_XS  + 64 * XS_P * 2)            // 51200
#define O_VT   (O_QKV + 64 * QKV_P * 2)           // +100352
#define O_SF   (O_VT  + NHEADS * HD * VT_P * 2)   // +61440
#define O_PB   (O_SF  + 64 * SF_P * 4)            // +17408
#define O_BQ   (O_PB  + 64 * PB_P * 2)            // +10240
#define O_BP   (O_BQ  + CH3 * 4)                  // +4608
#define SMEM_BYTES (O_BP + CH * 4)                // = 246784 B (< 320KB/WG)

static __device__ inline unsigned short f32_to_bf16(float f) {
  unsigned int u = __float_as_uint(f);
  unsigned int r = (u + 0x7FFFu + ((u >> 16) & 1u)) >> 16;   // RNE
  return (unsigned short)r;
}

static __device__ inline unsigned int pack2_bf16(float a, float b) {
  return (unsigned int)f32_to_bf16(a) | ((unsigned int)f32_to_bf16(b) << 16);
}

static __device__ inline v8f wmma_bf16(v16bf a, v16bf b, v8f c) {
  return __builtin_amdgcn_wmma_f32_16x16x32_bf16(
      /*neg_a=*/false, a, /*neg_b=*/false, b,
      /*c_mod=*/(short)0, c, /*reuse_a=*/false, /*reuse_b=*/false);
}

// A-fragment: row-major source row; CDNA5 interleaved 8-element K-chunks.
static __device__ inline v16bf ldA(const unsigned short* row, int k0, int hi) {
  union { v16bf v; v8bf h[2]; } u;
  const unsigned short* p = row + k0 + hi * 8;
  u.h[0] = *reinterpret_cast<const v8bf*>(p);        // K = k0+hi*8 .. +8
  u.h[1] = *reinterpret_cast<const v8bf*>(p + 16);   // K = k0+16+hi*8 .. +8
  return u.v;
}

// B-fragment: "column source" row (row n of B^T); same interleaved K-chunks as A.
static __device__ inline v16bf ldB(const unsigned short* row, int k0, int hi) {
  union { v16bf v; v8bf h[2]; } u;
  const unsigned short* p = row + k0 + hi * 8;
  u.h[0] = *reinterpret_cast<const v8bf*>(p);
  u.h[1] = *reinterpret_cast<const v8bf*>(p + 16);
  return u.v;
}

// --- weight transpose + bf16 convert: W (rows x cols) -> WT (cols x rows) ---
__global__ void cvt_transpose_bf16(const float* __restrict__ W,
                                   unsigned short* __restrict__ WT,
                                   int rows, int cols) {
  int idx = blockIdx.x * blockDim.x + threadIdx.x;
  if (idx < rows * cols) {
    int r = idx / cols, c = idx - r * cols;
    WT[(size_t)c * rows + r] = f32_to_bf16(W[idx]);
  }
}

__global__ __launch_bounds__(256)
void fused_window_attn(const float* __restrict__ x,
                       const float* __restrict__ bqkv,
                       const float* __restrict__ bproj,
                       const unsigned short* __restrict__ WqkvT,   // 1152 x 384 bf16
                       const unsigned short* __restrict__ WprojT,  // 384 x 384 bf16
                       float* __restrict__ out) {
  extern __shared__ unsigned char smem[];
  unsigned short* xs  = (unsigned short*)(smem + O_XS);   // 64 x XS_P
  unsigned short* ob  = xs;                               // alias: O staging
  unsigned short* qkv = (unsigned short*)(smem + O_QKV);  // 64 x QKV_P
  unsigned short* vt  = (unsigned short*)(smem + O_VT);   // (12*32) x VT_P
  float*          sf  = (float*)(smem + O_SF);            // 64 x SF_P
  unsigned short* pb  = (unsigned short*)(smem + O_PB);   // 64 x PB_P
  float*          bq  = (float*)(smem + O_BQ);
  float*          bp  = (float*)(smem + O_BP);

  const int tid  = threadIdx.x;
  const int lane = tid & 31;
  const int wave = tid >> 5;         // 0..7
  const int l15  = lane & 15;
  const int hi   = lane >> 4;        // 0/1: K-chunk select (A/B), M-half (C/D)
  const int b    = blockIdx.x;

  // ---- Phase A: stage x window as bf16 (zero-padded to 64 rows) + biases ----
  {
    unsigned int* xz = (unsigned int*)xs;
    for (int i = tid; i < 64 * XS_P / 2; i += 256) xz[i] = 0u;
    const float4* xb4 = (const float4*)(x + (size_t)b * SEQ * CH);
    // 49*384 floats = 4704 float4; dest offset m*XS_P + c (c = 4*quad in row)
    for (int i = tid; i < SEQ * CH / 4; i += 256) {
      int m  = i / (CH / 4);
      int c4 = (i - m * (CH / 4)) * 4;
      float4 v = xb4[i];
      unsigned long long w =
          (unsigned long long)pack2_bf16(v.x, v.y) |
          ((unsigned long long)pack2_bf16(v.z, v.w) << 32);
      *(unsigned long long*)(xs + m * XS_P + c4) = w;
    }
    for (int i = tid; i < CH3; i += 256) bq[i] = bqkv[i];
    for (int i = tid; i < CH;  i += 256) bp[i] = bproj[i];
  }
  __syncthreads();

  // ---- Phase B: QKV GEMM (64x384 @ 384x1152). 72 N-tiles over 8 waves. ----
  for (int nt = wave; nt < CH3 / 16; nt += 8) {
    const int n0  = nt * 16;
    const int col = n0 + l15;
    const unsigned short* wrow = WqkvT + (size_t)col * CH;  // B column source
    v8f acc0 = {}, acc1 = {}, acc2 = {}, acc3 = {};
#pragma unroll
    for (int ks = 0; ks < CH / 32; ++ks) {
      const int k0 = ks * 32;
      if (ks + 1 < CH / 32) __builtin_prefetch(wrow + k0 + 32, 0, 1);
      v16bf bf = ldB(wrow, k0, hi);
      v16bf a0 = ldA(xs + ( 0 + l15) * XS_P, k0, hi);
      v16bf a1 = ldA(xs + (16 + l15) * XS_P, k0, hi);
      v16bf a2 = ldA(xs + (32 + l15) * XS_P, k0, hi);
      v16bf a3 = ldA(xs + (48 + l15) * XS_P, k0, hi);
      acc0 = wmma_bf16(a0, bf, acc0);
      acc1 = wmma_bf16(a1, bf, acc1);
      acc2 = wmma_bf16(a2, bf, acc2);
      acc3 = wmma_bf16(a3, bf, acc3);
    }
    const float bias = bq[col];
    if (col < 2 * CH) {
      // Q | K region: scattered row-major stores (stride = QKV_P per M).
      auto store_qk = [&](v8f c, int m0) {
#pragma unroll
        for (int r = 0; r < 8; ++r)
          qkv[(m0 + hi * 8 + r) * QKV_P + col] = f32_to_bf16(c[r] + bias);
      };
      store_qk(acc0, 0); store_qk(acc1, 16); store_qk(acc2, 32); store_qk(acc3, 48);
    } else {
      // V region, stored transposed: lane's 8 values contiguous along seq.
      auto store_vt = [&](v8f c, int m0) {
        uint4 q;
        q.x = pack2_bf16(c[0] + bias, c[1] + bias);
        q.y = pack2_bf16(c[2] + bias, c[3] + bias);
        q.z = pack2_bf16(c[4] + bias, c[5] + bias);
        q.w = pack2_bf16(c[6] + bias, c[7] + bias);
        *(uint4*)(vt + (col - 2 * CH) * VT_P + m0 + hi * 8) = q;
      };
      store_vt(acc0, 0); store_vt(acc1, 16); store_vt(acc2, 32); store_vt(acc3, 48);
    }
  }
  __syncthreads();

  // ---- Phase C: per-head attention (heads sequential, waves tile S and O) ----
  const float alpha = 0.1178511301977579f;  // (hd^-0.5) / TEMPERATURE
  for (int h = 0; h < NHEADS; ++h) {
    // S = Q_h K_h^T : 4x4 tiles, K=hd=32 -> one WMMA per tile.
    for (int t = wave; t < 16; t += 8) {
      const int m0 = (t >> 2) * 16, n0 = (t & 3) * 16;
      v16bf a  = ldA(qkv + (m0 + l15) * QKV_P + h * HD, 0, hi);
      v16bf kb = ldB(qkv + (n0 + l15) * QKV_P + CH + h * HD, 0, hi);
      v8f c = {};
      c = wmma_bf16(a, kb, c);
#pragma unroll
      for (int r = 0; r < 8; ++r)
        sf[(m0 + hi * 8 + r) * SF_P + n0 + l15] = c[r];
    }
    __syncthreads();
    // Row softmax over 49 valid columns (cols >=49 zeroed in P).
    if (tid < 64) {
      float mx = -3.4e38f;
      for (int cc = 0; cc < SEQ; ++cc) mx = fmaxf(mx, sf[tid * SF_P + cc]);
      float sum = 0.0f;
      for (int cc = 0; cc < SEQ; ++cc) {
        float e = __expf((sf[tid * SF_P + cc] - mx) * alpha);
        sf[tid * SF_P + cc] = e;
        sum += e;
      }
      float inv = 1.0f / sum;
      for (int cc = 0; cc < 64; cc += 2) {
        float p0 = (cc     < SEQ) ? sf[tid * SF_P + cc]     * inv : 0.0f;
        float p1 = (cc + 1 < SEQ) ? sf[tid * SF_P + cc + 1] * inv : 0.0f;
        *(unsigned int*)(pb + tid * PB_P + cc) = pack2_bf16(p0, p1);
      }
    }
    __syncthreads();
    // O_h = P (64x64) @ V_h (64x32): 4x2 tiles, one per wave, K-loop of 2.
    {
      const int m0 = (wave >> 1) * 16, n0 = (wave & 1) * 16;
      v8f c = {};
#pragma unroll
      for (int ks = 0; ks < 2; ++ks) {
        v16bf a  = ldA(pb + (m0 + l15) * PB_P, ks * 32, hi);
        v16bf bv = ldB(vt + (h * HD + n0 + l15) * VT_P, ks * 32, hi);
        c = wmma_bf16(a, bv, c);
      }
#pragma unroll
      for (int r = 0; r < 8; ++r)
        ob[(m0 + hi * 8 + r) * XS_P + h * HD + n0 + l15] = f32_to_bf16(c[r]);
    }
    __syncthreads();
  }

  // ---- Phase D: projection GEMM (64x384 @ 384x384) + bias, write rows<49 ----
  float* outb = out + (size_t)b * SEQ * CH;
  for (int nt = wave; nt < CH / 16; nt += 8) {
    const int n0 = nt * 16;
    const unsigned short* wrow = WprojT + (size_t)(n0 + l15) * CH;
    v8f acc0 = {}, acc1 = {}, acc2 = {}, acc3 = {};
#pragma unroll
    for (int ks = 0; ks < CH / 32; ++ks) {
      const int k0 = ks * 32;
      if (ks + 1 < CH / 32) __builtin_prefetch(wrow + k0 + 32, 0, 1);
      v16bf bf = ldB(wrow, k0, hi);
      v16bf a0 = ldA(ob + ( 0 + l15) * XS_P, k0, hi);
      v16bf a1 = ldA(ob + (16 + l15) * XS_P, k0, hi);
      v16bf a2 = ldA(ob + (32 + l15) * XS_P, k0, hi);
      v16bf a3 = ldA(ob + (48 + l15) * XS_P, k0, hi);
      acc0 = wmma_bf16(a0, bf, acc0);
      acc1 = wmma_bf16(a1, bf, acc1);
      acc2 = wmma_bf16(a2, bf, acc2);
      acc3 = wmma_bf16(a3, bf, acc3);
    }
    const float bias = bp[n0 + l15];
    auto store_out = [&](v8f c, int m0) {
#pragma unroll
      for (int r = 0; r < 8; ++r) {
        int m = m0 + hi * 8 + r;
        if (m < SEQ) outb[m * CH + n0 + l15] = c[r] + bias;
      }
    };
    store_out(acc0, 0); store_out(acc1, 16); store_out(acc2, 32); store_out(acc3, 48);
  }
}

extern "C" void kernel_launch(void* const* d_in, const int* in_sizes, int n_in,
                              void* d_out, int out_size, void* d_ws, size_t ws_size,
                              hipStream_t stream) {
  (void)n_in; (void)out_size; (void)ws_size;
  const float* x     = (const float*)d_in[0];
  const float* Wqkv  = (const float*)d_in[1];
  const float* bqkv  = (const float*)d_in[2];
  const float* Wproj = (const float*)d_in[3];
  const float* bproj = (const float*)d_in[4];
  float* out = (float*)d_out;

  const int B = in_sizes[0] / (SEQ * CH);   // 8192

  unsigned short* wqkvT  = (unsigned short*)d_ws;            // 1152*384 bf16
  unsigned short* wprojT = wqkvT + (size_t)CH3 * CH;         // 384*384 bf16

  const int nq = CH * CH3;   // 442368
  const int np = CH * CH;    // 147456
  cvt_transpose_bf16<<<(nq + 255) / 256, 256, 0, stream>>>(Wqkv,  wqkvT,  CH, CH3);
  cvt_transpose_bf16<<<(np + 255) / 256, 256, 0, stream>>>(Wproj, wprojT, CH, CH);

  fused_window_attn<<<B, 256, SMEM_BYTES, stream>>>(x, bqkv, bproj, wqkvT, wprojT, out);
}
